// TritonVisionAttention_56521769615494
// MI455X (gfx1250) — matrix operand order repaired and checked by reading, not compiled, not run on hardware
//
#include <hip/hip_runtime.h>
#include <hip/hip_bf16.h>
#include <math.h>

// MI455X gfx1250: wave32, WMMA 16x16x32 f16 -> f32 accum.
// One block (128 threads = 4 waves) per (window, head) tile.
// Memory-bound problem (~0.68 GB traffic, ~10.7 GFLOP): single-pass
// load -> RoPE -> LDS(f16) -> WMMA QK^T -> softmax -> WMMA PV -> store.

#define TOTAL_SEQ   32768
#define NUM_HEADS   16
#define HEAD_DIM    80
#define HALF_DIM    40
#define WINDOW      64
#define NUM_WINDOWS (TOTAL_SEQ / WINDOW)
#define KPAD        96              // HEAD_DIM padded to 3 * 32 for WMMA k-chunks
#define QK_STRIDE   104             // 96 + 8 halves: 208 B rows, 16B-aligned, bank-spread
#define VT_STRIDE   72              // 64 + 8 halves: 144 B rows, 16B-aligned
#define P_STRIDE    72

typedef __attribute__((ext_vector_type(8)))  _Float16 v8h;
typedef __attribute__((ext_vector_type(16))) _Float16 v16h;
typedef __attribute__((ext_vector_type(8)))  float    v8f;

// A-fragment (16x32 f16, M x K) from a row-major LDS tile.
// Lane L<16: M=L, halves = K[kb+0..7], K[kb+16..23]
// Lane L>=16: M=L-16, halves = K[kb+8..15], K[kb+24..31]
__device__ __forceinline__ v16h load_frag_a(const _Float16* base, int stride,
                                            int m0, int kb, int lane) {
    const int mrow = lane & 15;
    const int hi   = (lane >> 4) & 1;
    const _Float16* p = base + (m0 + mrow) * stride + kb + hi * 8;
    v8h lo  = *(const v8h*)(p);
    v8h hi8 = *(const v8h*)(p + 16);
    v16h r;
#pragma unroll
    for (int i = 0; i < 8; ++i) { r[i] = lo[i]; r[8 + i] = hi8[i]; }
    return r;
}

// B-fragment (32x16 f16, K x N) where B[k][n] = T[n][k] for a row-major
// LDS tile T (i.e. T holds N rows of K values; QK^T: T=K-tile, PV: T=V^T).
// Lane L<16: N=L, halves = T[nb+L][kb+0..7], T[nb+L][kb+16..23]
// Lane L>=16: N=L-16, halves = T[nb+L-16][kb+8..15], T[nb+L-16][kb+24..31]
__device__ __forceinline__ v16h load_frag_b(const _Float16* base, int stride,
                                            int nb, int kb, int lane) {
    const int col = lane & 15;
    const int hi  = (lane >> 4) & 1;
    const _Float16* p = base + (nb + col) * stride + kb + hi * 8;
    v8h a = *(const v8h*)(p);
    v8h b = *(const v8h*)(p + 16);
    v16h r;
#pragma unroll
    for (int i = 0; i < 8; ++i) { r[i] = a[i]; r[8 + i] = b[i]; }
    return r;
}

__global__ void __launch_bounds__(128)
windowed_rope_attn_kernel(const float* __restrict__ q,
                          const float* __restrict__ k,
                          const float* __restrict__ v,
                          const float* __restrict__ cosp,
                          const float* __restrict__ sinp,
                          float* __restrict__ out) {
    __shared__ __align__(16) _Float16 Qs[WINDOW * QK_STRIDE];   // 13312 B
    __shared__ __align__(16) _Float16 Ks[WINDOW * QK_STRIDE];   // 13312 B
    __shared__ __align__(16) _Float16 Vt[HEAD_DIM * VT_STRIDE]; // 11520 B
    __shared__ __align__(16) _Float16 Ps[WINDOW * P_STRIDE];    //  9216 B

    const int t = threadIdx.x;       // 0..127
    const int w = blockIdx.x;        // window
    const int h = blockIdx.y;        // head
    const float scale = 0.11180339887498949f; // 1/sqrt(80)

    // ---------------- Load + RoPE + f16 convert into LDS ----------------
    {
        const int r = t >> 1;        // row in window, 0..63
        const int j = t & 1;         // half-of-row worker
        const size_t srow = (size_t)w * WINDOW + r;
        const float* qrow = q    + (srow * NUM_HEADS + h) * HEAD_DIM;
        const float* krow = k    + (srow * NUM_HEADS + h) * HEAD_DIM;
        const float* vrow = v    + (srow * NUM_HEADS + h) * HEAD_DIM;
        const float* crow = cosp + srow * HALF_DIM;
        const float* srow_ = sinp + srow * HALF_DIM;

#pragma unroll
        for (int i = 0; i < 20; ++i) {
            const int d = j * 20 + i;            // RoPE pair index 0..39
            const float c  = crow[d];
            const float s  = srow_[d];
            const float q0 = qrow[d], q1 = qrow[d + HALF_DIM];
            Qs[r * QK_STRIDE + d]            = (_Float16)((q0 * c - q1 * s) * scale);
            Qs[r * QK_STRIDE + d + HALF_DIM] = (_Float16)((q0 * s + q1 * c) * scale);
            const float k0 = krow[d], k1 = krow[d + HALF_DIM];
            Ks[r * QK_STRIDE + d]            = (_Float16)(k0 * c - k1 * s);
            Ks[r * QK_STRIDE + d + HALF_DIM] = (_Float16)(k0 * s + k1 * c);
        }
        // zero-fill K padding 80..95 so the third WMMA k-chunk is inert
#pragma unroll
        for (int i = 0; i < 8; ++i) {
            Qs[r * QK_STRIDE + HEAD_DIM + j * 8 + i] = (_Float16)0.0f;
            Ks[r * QK_STRIDE + HEAD_DIM + j * 8 + i] = (_Float16)0.0f;
        }
        // V transposed: Vt[d][n] = v[n][d]
#pragma unroll
        for (int i = 0; i < 40; ++i) {
            const int d = j * 40 + i;
            Vt[d * VT_STRIDE + r] = (_Float16)vrow[d];
        }
    }
    __syncthreads();

    // ---------------- Per-wave 16-row M-strip ----------------
    const int lane = t & 31;
    const int wv   = t >> 5;       // wave 0..3
    const int m0   = wv * 16;
    const int hi   = (lane >> 4) & 1;
    const int col  = lane & 15;

    // S = Qr * Kr^T : 4 N-tiles x 3 k-chunks of v_wmma_f32_16x16x32_f16
    v8f acc[4];
#pragma unroll
    for (int nt = 0; nt < 4; ++nt) {
        v8f a = {};
#pragma unroll
        for (int kb = 0; kb < KPAD; kb += 32) {
            v16h af = load_frag_a(Qs, QK_STRIDE, m0, kb, lane);
            v16h bf = load_frag_b(Ks, QK_STRIDE, nt * 16, kb, lane);
            a = __builtin_amdgcn_wmma_f32_16x16x32_f16(
                    false, af, false, bf, (short)0, a, false, false);
        }
        acc[nt] = a;
    }

    // ---------------- Softmax over the 64 columns of each row ----------------
    // D layout: VGPR r, lanes 0-15 -> row m0+r, lanes 16-31 -> row m0+r+8;
    // the 16 columns of a tile live across a 16-lane half, so xor-reductions
    // with masks 1,2,4,8 reduce over columns without crossing halves.
#pragma unroll
    for (int r = 0; r < 8; ++r) {
        float m = acc[0][r];
#pragma unroll
        for (int nt = 1; nt < 4; ++nt) m = fmaxf(m, acc[nt][r]);
#pragma unroll
        for (int mask = 8; mask >= 1; mask >>= 1)
            m = fmaxf(m, __shfl_xor(m, mask, 32));

        float sum = 0.0f;
#pragma unroll
        for (int nt = 0; nt < 4; ++nt) {
            const float e = __expf(acc[nt][r] - m);
            acc[nt][r] = e;
            sum += e;
        }
#pragma unroll
        for (int mask = 8; mask >= 1; mask >>= 1)
            sum += __shfl_xor(sum, mask, 32);
        const float inv = 1.0f / sum;

        const int row = m0 + r + hi * 8;
#pragma unroll
        for (int nt = 0; nt < 4; ++nt)
            Ps[row * P_STRIDE + nt * 16 + col] = (_Float16)(acc[nt][r] * inv);
    }
    // Ps rows [m0, m0+16) are written and read only by this wave; same-wave
    // LDS ops are in-order (DScnt), so no cross-wave barrier is required.

    // ---------------- O = P * V : 5 D-tiles x 2 k-chunks ----------------
#pragma unroll
    for (int dt = 0; dt < 5; ++dt) {
        v8f o = {};
#pragma unroll
        for (int kb = 0; kb < 64; kb += 32) {
            v16h af = load_frag_a(Ps, P_STRIDE, m0, kb, lane);
            v16h bf = load_frag_b(Vt, VT_STRIDE, dt * 16, kb, lane);
            o = __builtin_amdgcn_wmma_f32_16x16x32_f16(
                    false, af, false, bf, (short)0, o, false, false);
        }
#pragma unroll
        for (int r = 0; r < 8; ++r) {
            const size_t srow = (size_t)w * WINDOW + m0 + r + hi * 8;
            out[(srow * NUM_HEADS + h) * HEAD_DIM + dt * 16 + col] = o[r];
        }
    }
}

extern "C" void kernel_launch(void* const* d_in, const int* in_sizes, int n_in,
                              void* d_out, int out_size, void* d_ws, size_t ws_size,
                              hipStream_t stream) {
    (void)in_sizes; (void)n_in; (void)out_size; (void)d_ws; (void)ws_size;
    const float* q    = (const float*)d_in[0];
    const float* k    = (const float*)d_in[1];
    const float* v    = (const float*)d_in[2];
    const float* cosp = (const float*)d_in[3];
    const float* sinp = (const float*)d_in[4];
    // d_in[5] = cu_seqlens: windows are uniform (WINDOW=64); unused.
    float* out = (float*)d_out;

    dim3 grid(NUM_WINDOWS, NUM_HEADS, 1);   // 512 x 16 = 8192 blocks
    dim3 block(128, 1, 1);                  // 4 waves, one 16-row strip each
    windowed_rope_attn_kernel<<<grid, block, 0, stream>>>(q, k, v, cosp, sinp, out);
}
